// SilencioActivoNetwork_64175401337406
// MI455X (gfx1250) — compile-verified, matrix operand order checked by reading, not run on hardware
//
#include <hip/hip_runtime.h>
#include <hip/hip_bf16.h>
#include <math.h>
#include <stdint.h>

// ---------------------------------------------------------------------------
// CDNA5 (gfx1250, wave32) implementation.
// Polar factor via Newton-Schulz (bf16 WMMA gemms, f32 accumulate), fused
// ReLU / axpby epilogues, Gram gemm + softmax-entropy reduction, on-device
// gating, final small linear.
// NT gemm uses async global->LDS staging (ASYNCcnt) + double-buffered LDS
// + L2 prefetch; NN gemm (transpose staging) keeps the VGPR staging path.
// ---------------------------------------------------------------------------

typedef __attribute__((ext_vector_type(16))) __bf16 v16bf;
typedef __attribute__((ext_vector_type(8)))  float  v8f;

#define TILE_M 128
#define TILE_N 128
#define TILE_K 32
#define LDSTR  40      // bf16 elems per LDS row: 32 + 8 pad (80B, 16B aligned)

enum { EPI_STORE = 0, EPI_RELU = 1, EPI_AXPBY = 2 };

union Frag32B { v16bf v; uint4 q[2]; };
union Vec8BF  { uint4 q; __bf16 h[8]; };

// ---------------------------------------------------------------------------
// Async global -> LDS 16B copy (CDNA5 GLOBAL_LOAD_ASYNC_TO_LDS_B128).
// Builtin signature (from hipcc diagnostic): param0 = v4i in addrspace(1),
// param1 (LDS dest) expected in addrspace(3), plus two immediate ints.
// ---------------------------------------------------------------------------
#if defined(__HIP_DEVICE_COMPILE__) && defined(__gfx1250__)
  #if __has_builtin(__builtin_amdgcn_global_load_async_to_lds_b128)
    #define ASYNC_MODE 1
  #else
    #define ASYNC_MODE 2
  #endif
#else
  #define ASYNC_MODE 0
#endif

typedef int v4i_16B __attribute__((__vector_size__(4 * sizeof(int))));
typedef __attribute__((address_space(1))) v4i_16B* gas_v4i_ptr;
typedef __attribute__((address_space(3))) v4i_16B* lds_v4i_ptr;

__device__ __forceinline__ void async_copy_b128(const void* gsrc, void* ldst)
{
#if ASYNC_MODE == 1
    __builtin_amdgcn_global_load_async_to_lds_b128(
        (gas_v4i_ptr)(uintptr_t)gsrc,
        (lds_v4i_ptr)(uintptr_t)ldst,   // generic LDS addr: low 32 bits = offset
        0, 0);
#elif ASYNC_MODE == 2
    unsigned lds32 = (unsigned)(uintptr_t)ldst;
    asm volatile("global_load_async_to_lds_b128 %0, %1, off"
                 :: "v"(lds32), "v"(gsrc) : "memory");
#else
    *(uint4*)ldst = *(const uint4*)gsrc;
#endif
}

__device__ __forceinline__ void async_wait_all()
{
#if ASYNC_MODE == 1
  #if __has_builtin(__builtin_amdgcn_s_wait_asynccnt)
    __builtin_amdgcn_s_wait_asynccnt(0);
  #else
    asm volatile("s_wait_asynccnt 0x0" ::: "memory");
  #endif
#elif ASYNC_MODE == 2
    asm volatile("s_wait_asynccnt 0x0" ::: "memory");
#endif
}

// ---------------------------------------------------------------------------
// NT GEMM: C(f32, MxN) = A(bf16, MxK rm) * B(bf16, NxK rm)^T [+ epilogue]
// Double-buffered LDS, async staging. M,N mult of 128; K mult of 32.
// ---------------------------------------------------------------------------
template<int EPI>
__global__ __launch_bounds__(256)
void gemm_bf16_nt_kernel(const __bf16* __restrict__ A,
                         const __bf16* __restrict__ Bm,
                         float* __restrict__ C,
                         const float* __restrict__ Aux,
                         int M, int N, int K,
                         int lda, int ldb, int ldc,
                         float alpha, float beta)
{
    __shared__ __align__(16) __bf16 As[2][TILE_M][LDSTR];
    __shared__ __align__(16) __bf16 Bs[2][TILE_N][LDSTR];

    const int tid  = threadIdx.x;
    const int lane = tid & 31;
    const int wave = tid >> 5;     // 0..7
    const int wm   = wave >> 1;    // 0..3 -> 32 rows
    const int wn   = wave & 1;     // 0..1 -> 64 cols

    const int rowBase = blockIdx.y * TILE_M;
    const int colBase = blockIdx.x * TILE_N;

    const int laneHalf = lane >> 4;
    const int kh       = laneHalf << 3;
    const int lr       = lane & 15;

    v8f acc[2][4];
    #pragma unroll
    for (int i = 0; i < 2; ++i)
        #pragma unroll
        for (int j = 0; j < 4; ++j)
            #pragma unroll
            for (int e = 0; e < 8; ++e) acc[i][j][e] = 0.0f;

    const int nk = K / TILE_K;

    auto stage = [&](int p, int k0) {
        #pragma unroll
        for (int i = 0; i < 4; ++i) {
            int idx = i * 256 + tid;          // 0..1023
            int r   = idx >> 2;               // 0..255
            int cv  = (idx & 3) << 3;         // 0,8,16,24
            async_copy_b128(A + (size_t)(rowBase + r) * lda + (size_t)(k0 + cv),
                            &As[p][r][cv]);
            async_copy_b128(Bm + (size_t)(colBase + r) * ldb + (size_t)(k0 + cv),
                            &Bs[p][r][cv]);
        }
    };

    stage(0, 0);

    for (int kt = 0; kt < nk; ++kt) {
        const int p = kt & 1;
        async_wait_all();        // own async copies of buf[p] landed
        __syncthreads();         // -> everyone's copies landed; compute(kt-1) done
        if (kt + 1 < nk) stage(p ^ 1, (kt + 1) * TILE_K);
        if (kt + 2 < nk) {       // warm L2 two tiles ahead (global_prefetch_b8)
            int k2 = (kt + 2) * TILE_K + (tid & 1) * 16;
            int r2 = tid >> 1;   // 0..127
            __builtin_prefetch(A  + (size_t)(rowBase + r2) * lda + k2, 0, 0);
            __builtin_prefetch(Bm + (size_t)(colBase + r2) * ldb + k2, 0, 0);
        }

        Frag32B afrag[2], bfrag[4];
        #pragma unroll
        for (int mi = 0; mi < 2; ++mi) {
            int r = wm * 32 + mi * 16 + lr;
            afrag[mi].q[0] = *reinterpret_cast<const uint4*>(&As[p][r][kh]);
            afrag[mi].q[1] = *reinterpret_cast<const uint4*>(&As[p][r][kh + 16]);
        }
        #pragma unroll
        for (int ni = 0; ni < 4; ++ni) {
            int c = wn * 64 + ni * 16 + lr;
            bfrag[ni].q[0] = *reinterpret_cast<const uint4*>(&Bs[p][c][kh]);
            bfrag[ni].q[1] = *reinterpret_cast<const uint4*>(&Bs[p][c][kh + 16]);
        }
        #pragma unroll
        for (int mi = 0; mi < 2; ++mi)
            #pragma unroll
            for (int ni = 0; ni < 4; ++ni)
                acc[mi][ni] = __builtin_amdgcn_wmma_f32_16x16x32_bf16(
                    false, afrag[mi].v, false, bfrag[ni].v,
                    (short)0, acc[mi][ni], false, false);
    }

    #pragma unroll
    for (int mi = 0; mi < 2; ++mi) {
        #pragma unroll
        for (int ni = 0; ni < 4; ++ni) {
            #pragma unroll
            for (int r = 0; r < 8; ++r) {
                int row = rowBase + wm * 32 + mi * 16 + laneHalf * 8 + r;
                int col = colBase + wn * 64 + ni * 16 + lr;
                size_t o = (size_t)row * ldc + col;
                float v = acc[mi][ni][r];
                if (EPI == EPI_RELU)        C[o] = v > 0.0f ? v : 0.0f;
                else if (EPI == EPI_AXPBY)  C[o] = alpha * v + beta * Aux[o];
                else                        C[o] = v;
            }
        }
    }
}

// ---------------------------------------------------------------------------
// NN GEMM: C = A(bf16, MxK rm) * B(bf16, KxN rm); transpose while staging B.
// Used only for T*X inside Newton-Schulz.
// ---------------------------------------------------------------------------
template<int EPI>
__global__ __launch_bounds__(256)
void gemm_bf16_nn_kernel(const __bf16* __restrict__ A,
                         const __bf16* __restrict__ Bm,
                         float* __restrict__ C,
                         const float* __restrict__ Aux,
                         int M, int N, int K,
                         int lda, int ldb, int ldc,
                         float alpha, float beta)
{
    __shared__ __align__(16) __bf16 As[TILE_M][LDSTR];
    __shared__ __align__(16) __bf16 Bs[TILE_N][LDSTR];

    const int tid  = threadIdx.x;
    const int lane = tid & 31;
    const int wave = tid >> 5;
    const int wm   = wave >> 1;
    const int wn   = wave & 1;

    const int rowBase = blockIdx.y * TILE_M;
    const int colBase = blockIdx.x * TILE_N;

    const int laneHalf = lane >> 4;
    const int kh       = laneHalf << 3;
    const int lr       = lane & 15;

    v8f acc[2][4];
    #pragma unroll
    for (int i = 0; i < 2; ++i)
        #pragma unroll
        for (int j = 0; j < 4; ++j)
            #pragma unroll
            for (int e = 0; e < 8; ++e) acc[i][j][e] = 0.0f;

    for (int k0 = 0; k0 < K; k0 += TILE_K) {
        #pragma unroll
        for (int i = 0; i < 4; ++i) {
            int idx = i * 256 + tid;
            int r   = idx >> 2;
            int cv  = (idx & 3) << 3;
            uint4 d = *reinterpret_cast<const uint4*>(
                A + (size_t)(rowBase + r) * lda + (size_t)(k0 + cv));
            *reinterpret_cast<uint4*>(&As[r][cv]) = d;
        }
        // B is (K x N) row-major: transpose while staging into Bs[n][k].
        #pragma unroll
        for (int i = 0; i < 2; ++i) {
            int vidx = i * 256 + tid;     // 0..511
            int kk   = vidx >> 4;         // 0..31
            int nv   = (vidx & 15) << 3;  // 0..120
            Vec8BF d;
            d.q = *reinterpret_cast<const uint4*>(
                Bm + (size_t)(k0 + kk) * ldb + (size_t)(colBase + nv));
            #pragma unroll
            for (int j = 0; j < 8; ++j) Bs[nv + j][kk] = d.h[j];
        }
        __syncthreads();

        Frag32B afrag[2], bfrag[4];
        #pragma unroll
        for (int mi = 0; mi < 2; ++mi) {
            int r = wm * 32 + mi * 16 + lr;
            afrag[mi].q[0] = *reinterpret_cast<const uint4*>(&As[r][kh]);
            afrag[mi].q[1] = *reinterpret_cast<const uint4*>(&As[r][kh + 16]);
        }
        #pragma unroll
        for (int ni = 0; ni < 4; ++ni) {
            int c = wn * 64 + ni * 16 + lr;
            bfrag[ni].q[0] = *reinterpret_cast<const uint4*>(&Bs[c][kh]);
            bfrag[ni].q[1] = *reinterpret_cast<const uint4*>(&Bs[c][kh + 16]);
        }
        #pragma unroll
        for (int mi = 0; mi < 2; ++mi)
            #pragma unroll
            for (int ni = 0; ni < 4; ++ni)
                acc[mi][ni] = __builtin_amdgcn_wmma_f32_16x16x32_bf16(
                    false, afrag[mi].v, false, bfrag[ni].v,
                    (short)0, acc[mi][ni], false, false);
        __syncthreads();
    }

    #pragma unroll
    for (int mi = 0; mi < 2; ++mi) {
        #pragma unroll
        for (int ni = 0; ni < 4; ++ni) {
            #pragma unroll
            for (int r = 0; r < 8; ++r) {
                int row = rowBase + wm * 32 + mi * 16 + laneHalf * 8 + r;
                int col = colBase + wn * 64 + ni * 16 + lr;
                size_t o = (size_t)row * ldc + col;
                float v = acc[mi][ni][r];
                if (EPI == EPI_RELU)        C[o] = v > 0.0f ? v : 0.0f;
                else if (EPI == EPI_AXPBY)  C[o] = alpha * v + beta * Aux[o];
                else                        C[o] = v;
            }
        }
    }
}

// ---------------------------------------------------------------------------
// Elementwise helpers
// ---------------------------------------------------------------------------
__global__ void cvt_f32_bf16_kernel(const float* __restrict__ s,
                                    __bf16* __restrict__ d, size_t n)
{
    size_t i = (size_t)blockIdx.x * 256 + threadIdx.x;
    if (i < n) d[i] = (__bf16)s[i];
}

__global__ void copy_f32_kernel(const float* __restrict__ s,
                                float* __restrict__ d, size_t n)
{
    size_t i = (size_t)blockIdx.x * 256 + threadIdx.x;
    if (i < n) d[i] = s[i];
}

__global__ void gate_apply_kernel(const float* __restrict__ s,
                                  const float* __restrict__ gate,
                                  float* __restrict__ dF,
                                  __bf16* __restrict__ dB, size_t n)
{
    size_t i = (size_t)blockIdx.x * 256 + threadIdx.x;
    if (i < n) {
        float v = s[i] * (*gate);
        dF[i] = v;
        dB[i] = (__bf16)v;
    }
}

__global__ void set_scalar_kernel(float* p, float v)
{
    if (threadIdx.x == 0 && blockIdx.x == 0) *p = v;
}

// Row softmax + entropy contribution, accumulated into *S_acc.
__global__ __launch_bounds__(256)
void softmax_entropy_kernel(const float* __restrict__ G, int n,
                            float* __restrict__ S_acc)
{
    __shared__ float red[256];
    const float* g = G + (size_t)blockIdx.x * n;
    const int tid = threadIdx.x;

    float m = -3.402823466e38f;
    for (int i = tid; i < n; i += 256) m = fmaxf(m, g[i]);
    red[tid] = m; __syncthreads();
    for (int s = 128; s > 0; s >>= 1) {
        if (tid < s) red[tid] = fmaxf(red[tid], red[tid + s]);
        __syncthreads();
    }
    m = red[0]; __syncthreads();

    float z = 0.0f;
    for (int i = tid; i < n; i += 256) z += __expf(g[i] - m);
    red[tid] = z; __syncthreads();
    for (int s = 128; s > 0; s >>= 1) {
        if (tid < s) red[tid] += red[tid + s];
        __syncthreads();
    }
    z = red[0]; __syncthreads();
    const float inv = 1.0f / z;

    float e = 0.0f;
    for (int i = tid; i < n; i += 256) {
        float p = __expf(g[i] - m) * inv;
        e += p * __logf(p + 1e-12f);
    }
    red[tid] = e; __syncthreads();
    for (int s = 128; s > 0; s >>= 1) {
        if (tid < s) red[tid] += red[tid + s];
        __syncthreads();
    }
    if (tid == 0) atomicAdd(S_acc, -red[0]);
}

__global__ void finalize_kernel(const float* __restrict__ S_acc,
                                const float* __restrict__ phi,
                                float* __restrict__ delta_out,
                                float* __restrict__ gate)
{
    const float LOG5 = 1.6094379124341003f;
    const float EPS_BASE = 0.6931471805599453f * (1.0f / 7.3f) * (1.0f / 7.3f);
    float ds = *S_acc - LOG5;
    *delta_out = ds;
    float xi = fabsf(__cosf(phi[0]) * __sinf(phi[1]) * __cosf(phi[2]));
    float eps_c = EPS_BASE * (1.0f + xi);
    *gate = (eps_c - ds > 0.0f) ? 1.0f : 0.0f;
}

// logits = h @ Wout^T + bout  (4096 x 512 x 10: negligible, plain VALU)
__global__ void final_linear_kernel(const float* __restrict__ h,
                                    const float* __restrict__ Wo,
                                    const float* __restrict__ bo,
                                    float* __restrict__ out,
                                    int B, int K, int O)
{
    int idx = blockIdx.x * blockDim.x + threadIdx.x;
    if (idx >= B * O) return;
    int r = idx / O, o = idx % O;
    const float* hr = h + (size_t)r * K;
    const float* wr = Wo + (size_t)o * K;
    float s = bo[o];
    for (int k = 0; k < K; ++k) s += hr[k] * wr[k];
    out[idx] = s;
}

// ---------------------------------------------------------------------------
// Host-side orchestration
// ---------------------------------------------------------------------------
static inline int grid1(size_t n) { return (int)((n + 255) / 256); }

static void run_gemm_nt(int epi, const __bf16* A, const __bf16* B, float* C,
                        const float* Aux, int M, int N, int K,
                        int lda, int ldb, int ldc,
                        float alpha, float beta, hipStream_t st)
{
    dim3 g(N / TILE_N, M / TILE_M), b(256);
    if (epi == EPI_STORE)
        gemm_bf16_nt_kernel<EPI_STORE><<<g, b, 0, st>>>(A, B, C, Aux, M, N, K, lda, ldb, ldc, alpha, beta);
    else if (epi == EPI_RELU)
        gemm_bf16_nt_kernel<EPI_RELU><<<g, b, 0, st>>>(A, B, C, Aux, M, N, K, lda, ldb, ldc, alpha, beta);
    else
        gemm_bf16_nt_kernel<EPI_AXPBY><<<g, b, 0, st>>>(A, B, C, Aux, M, N, K, lda, ldb, ldc, alpha, beta);
}

static void run_gemm_nn_axpby(const __bf16* A, const __bf16* B, float* C,
                              const float* Aux, int M, int N, int K,
                              int lda, int ldb, int ldc,
                              float alpha, float beta, hipStream_t st)
{
    dim3 g(N / TILE_N, M / TILE_M), b(256);
    gemm_bf16_nn_kernel<EPI_AXPBY><<<g, b, 0, st>>>(A, B, C, Aux, M, N, K, lda, ldb, ldc, alpha, beta);
}

extern "C" void kernel_launch(void* const* d_in, const int* in_sizes, int n_in,
                              void* d_out, int out_size, void* d_ws, size_t ws_size,
                              hipStream_t stream)
{
    (void)in_sizes; (void)n_in; (void)out_size; (void)ws_size;

    const float* x      = (const float*)d_in[0];
    const float* W[3]   = {(const float*)d_in[1], (const float*)d_in[2], (const float*)d_in[3]};
    const float* phi[3] = {(const float*)d_in[4], (const float*)d_in[5], (const float*)d_in[6]};
    const float* Wout   = (const float*)d_in[7];
    const float* bout   = (const float*)d_in[8];
    float* out          = (float*)d_out;

    const int Bsz = 4096;
    const int dims[4] = {2048, 2048, 2048, 512};
    const int NS_ITERS = 26;   // sigma_min ~ 2e-4 grows x1.5/iter -> ~26 to converge

    char* ws = (char*)d_ws;
    size_t off = 0;
    auto carve = [&](size_t bytes) -> char* {
        char* p = ws + off;
        off += (bytes + 255) & ~(size_t)255;
        return p;
    };
    float*  S_acc = (float*)carve(4);
    float*  gate  = (float*)carve(4);
    float*  Xa    = (float*)carve((size_t)2048 * 2048 * 4);
    float*  Xb32  = (float*)carve((size_t)2048 * 2048 * 4);
    float*  T32   = (float*)carve((size_t)2048 * 2048 * 4);
    __bf16* X16   = (__bf16*)carve((size_t)2048 * 2048 * 2);
    __bf16* T16   = (__bf16*)carve((size_t)2048 * 2048 * 2);
    __bf16* D16   = (__bf16*)carve((size_t)2048 * 2048 * 2);
    __bf16* hin16 = (__bf16*)carve((size_t)Bsz * 2048 * 2);
    float*  hn32  = (float*)carve((size_t)Bsz * 2048 * 4);
    __bf16* hn16  = (__bf16*)carve((size_t)Bsz * 2048 * 2);
    float*  hg32  = (float*)carve((size_t)Bsz * 2048 * 4);
    float*  G32   = (float*)carve((size_t)Bsz * Bsz * 4);

    {   // h0 = bf16(x)
        size_t n = (size_t)Bsz * dims[0];
        cvt_f32_bf16_kernel<<<grid1(n), 256, 0, stream>>>(x, hin16, n);
    }

    for (int l = 0; l < 3; ++l) {
        const int Mp = dims[l + 1];
        const int Kp = dims[l];
        const size_t nW = (size_t)Mp * Kp;

        // ---- polar factor via Newton-Schulz: X <- 1.5X - 0.5 (X X^T) X ----
        copy_f32_kernel<<<grid1(nW), 256, 0, stream>>>(W[l], Xa, nW);
        float* Xc = Xa;
        float* Xn = Xb32;
        for (int it = 0; it < NS_ITERS; ++it) {
            cvt_f32_bf16_kernel<<<grid1(nW), 256, 0, stream>>>(Xc, X16, nW);
            run_gemm_nt(EPI_STORE, X16, X16, T32, nullptr,
                        Mp, Mp, Kp, Kp, Kp, Mp, 0.0f, 0.0f, stream);
            cvt_f32_bf16_kernel<<<grid1((size_t)Mp * Mp), 256, 0, stream>>>(
                T32, T16, (size_t)Mp * Mp);
            run_gemm_nn_axpby(T16, X16, Xn, Xc,
                              Mp, Kp, Mp, Mp, Kp, Kp, -0.5f, 1.5f, stream);
            float* t = Xc; Xc = Xn; Xn = t;
        }
        cvt_f32_bf16_kernel<<<grid1(nW), 256, 0, stream>>>(Xc, D16, nW);

        // ---- forward: hn = relu(h @ D^T) ----
        run_gemm_nt(EPI_RELU, hin16, D16, hn32, nullptr,
                    Bsz, Mp, Kp, Kp, Kp, Mp, 0.0f, 0.0f, stream);
        {
            size_t n = (size_t)Bsz * Mp;
            cvt_f32_bf16_kernel<<<grid1(n), 256, 0, stream>>>(hn32, hn16, n);
        }

        // ---- Gram + entropy gate ----
        run_gemm_nt(EPI_STORE, hn16, hn16, G32, nullptr,
                    Bsz, Bsz, Mp, Mp, Mp, Bsz, 0.0f, 0.0f, stream);
        set_scalar_kernel<<<1, 1, 0, stream>>>(S_acc, 0.0f);
        softmax_entropy_kernel<<<Bsz, 256, 0, stream>>>(G32, Bsz, S_acc);
        finalize_kernel<<<1, 1, 0, stream>>>(S_acc, phi[l],
                                             out + (size_t)Bsz * 10 + l, gate);

        // ---- apply gate; produce next-layer input ----
        {
            size_t n = (size_t)Bsz * Mp;
            gate_apply_kernel<<<grid1(n), 256, 0, stream>>>(hn32, gate, hg32, hin16, n);
        }
    }

    final_linear_kernel<<<grid1((size_t)Bsz * 10), 256, 0, stream>>>(
        hg32, Wout, bout, out, Bsz, dims[3], 10);
}
